// ScaledDotProductAttention_7584912245195
// MI455X (gfx1250) — matrix operand order repaired and checked by reading, not compiled, not run on hardware
//
#include <hip/hip_runtime.h>
#include <stdint.h>

typedef __attribute__((ext_vector_type(16))) _Float16 v16h;
typedef __attribute__((ext_vector_type(8)))  _Float16 v8h;
typedef __attribute__((ext_vector_type(2)))  _Float16 v2h;
typedef __attribute__((ext_vector_type(8)))  float    v8f;

#define BB 2
#define HH 16
#define SS 2048
#define DD 64
#define QTILE 16
#define NWAVES 4
#define FSTRIDE 2050   // f32 score row stride (even -> aligned float2, bank-safe)
#define HSTRIDE 2056   // f16 prob row stride (+8 halves = 16B pad -> conflict-free b128)
#define SCALE 0.125f   // 1/sqrt(64)

__global__ __launch_bounds__(NWAVES * 32)
void sdpa_fused_kernel(const float* __restrict__ Q,
                       const float* __restrict__ K,
                       const float* __restrict__ V,
                       const unsigned char* __restrict__ mask,
                       float* __restrict__ ctx_out,
                       float* __restrict__ attn_out) {
    extern __shared__ float smem[];
    float*    smf = smem;                                   // [16][FSTRIDE] f32 scores/exp
    _Float16* smh = (_Float16*)(smem + QTILE * FSTRIDE);    // [16][HSTRIDE] f16 probs

    const int q0   = blockIdx.x * QTILE;
    const int h    = blockIdx.y;
    const int b    = blockIdx.z;
    const int tid  = threadIdx.x;
    const int wave = tid >> 5;
    const int lane = tid & 31;

    const int m      = lane & 15;
    const int hsel8  = (lane < 16) ? 0 : 8;
    const int hsel16 = (lane < 16) ? 0 : 16;

    const size_t bh        = (size_t)b * HH + h;
    const size_t qkv_base  = bh * SS * DD;
    const size_t mask_base = bh * (size_t)SS * SS + (size_t)q0 * SS;
    const size_t attn_base = (bh * SS + (size_t)q0) * SS;

    // ---------------- Phase 1: scores = mask( (Q*scale) @ K^T ) -> f32 LDS ----------
    v16h a0, a1;   // Q tile 16x64 as two 16x32 f16 A-fragments (scale folded in)
    {
        const float* qrow = Q + qkv_base + (size_t)(q0 + m) * DD;
#pragma unroll
        for (int i = 0; i < 16; ++i) {
            const int kk = i + hsel16;   // 0..31
            a0[i] = (_Float16)(qrow[kk]      * SCALE);
            a1[i] = (_Float16)(qrow[32 + kk] * SCALE);
        }
    }

    for (int kt = wave; kt < SS / 16; kt += NWAVES) {
        const int n0 = kt * 16;
        const float* krow = K + qkv_base + (size_t)(n0 + m) * DD;
        if (kt + NWAVES < SS / 16)
            __builtin_prefetch(krow + (size_t)NWAVES * 16 * DD, 0, 0);
        v16h b0, b1;   // B = K^T tile: (Kdim=d, N=key) = K[key][d], contiguous along d
#pragma unroll
        for (int i = 0; i < 16; ++i) {
            b0[i] = (_Float16)krow[hsel16 + i];
            b1[i] = (_Float16)krow[32 + hsel16 + i];
        }
        v8f c = {};
        c = __builtin_amdgcn_wmma_f32_16x16x32_f16(false, a0, false, b0,
                                                   (short)0, c, false, false);
        c = __builtin_amdgcn_wmma_f32_16x16x32_f16(false, a1, false, b1,
                                                   (short)0, c, false, false);
        const int col = n0 + m;
#pragma unroll
        for (int i = 0; i < 8; ++i) {
            const int row = i + hsel8;
            float s = c[i];
            if (mask[mask_base + (size_t)row * SS + col]) s = -1e9f;
            smf[row * FSTRIDE + col] = s;
        }
    }
    __syncthreads();

    // ---------------- Phase 2: softmax rows; attn out (float2) + f16 probs to LDS ---
    for (int r = wave * 4; r < wave * 4 + 4; ++r) {
        float* srow = smf + r * FSTRIDE;
        float mx = -1e30f;
        for (int cb = lane * 2; cb < SS; cb += 64) {
            const float2 s2 = *(const float2*)(srow + cb);
            mx = fmaxf(mx, fmaxf(s2.x, s2.y));
        }
#pragma unroll
        for (int off = 16; off; off >>= 1) mx = fmaxf(mx, __shfl_xor(mx, off, 32));
        float sum = 0.0f;
        for (int cb = lane * 2; cb < SS; cb += 64) {
            const float2 s2 = *(const float2*)(srow + cb);
            const float e0 = __expf(s2.x - mx);
            const float e1 = __expf(s2.y - mx);
            sum += e0 + e1;
            *(float2*)(srow + cb) = make_float2(e0, e1);
        }
#pragma unroll
        for (int off = 16; off; off >>= 1) sum += __shfl_xor(sum, off, 32);
        const float inv = 1.0f / sum;
        float*    arow = attn_out + attn_base + (size_t)r * SS;
        _Float16* hrow = smh + r * HSTRIDE;
        for (int cb = lane * 2; cb < SS; cb += 64) {
            const float2 e2 = *(const float2*)(srow + cb);
            const float p0 = e2.x * inv;
            const float p1 = e2.y * inv;
            *(float2*)(arow + cb) = make_float2(p0, p1);       // 8B coalesced global
            *(v2h*)(hrow + cb)    = (v2h){(_Float16)p0, (_Float16)p1};
        }
    }
    __syncthreads();

    // ---------------- Phase 3: context = P @ V, wave owns a 16-wide d tile ----------
    const int d0 = wave * 16;
    const _Float16* prow = smh + m * HSTRIDE;   // row-major f16 probs, row m
    v8f acc = {};
    for (int k0 = 0; k0 < SS; k0 += 32) {
        // A fragment: two contiguous 16B runs per lane -> ds_load_b128 x2
        const v8h lo = *(const v8h*)(prow + k0 + hsel16);
        const v8h hi = *(const v8h*)(prow + k0 + hsel16 + 8);
        const v16h a = __builtin_shufflevector(lo, hi, 0, 1, 2, 3, 4, 5, 6, 7,
                                               8, 9, 10, 11, 12, 13, 14, 15);
        v16h bv;   // B = V tile: (Kdim=key, N=d) = V[key][d]
        const float* vcol = V + qkv_base + (size_t)(k0 + hsel16) * DD + d0 + m;
#pragma unroll
        for (int i = 0; i < 16; ++i)
            bv[i] = (_Float16)vcol[(size_t)i * DD];
        if (k0 + 32 < SS)
            __builtin_prefetch(vcol + (size_t)32 * DD, 0, 0);
        acc = __builtin_amdgcn_wmma_f32_16x16x32_f16(false, a, false, bv,
                                                     (short)0, acc, false, false);
    }
#pragma unroll
    for (int i = 0; i < 8; ++i) {
        const int row = i + hsel8;
        ctx_out[qkv_base + (size_t)(q0 + row) * DD + d0 + m] = acc[i];
    }
}

extern "C" void kernel_launch(void* const* d_in, const int* in_sizes, int n_in,
                              void* d_out, int out_size, void* d_ws, size_t ws_size,
                              hipStream_t stream) {
    (void)in_sizes; (void)n_in; (void)d_ws; (void)ws_size; (void)out_size;
    const float* Q = (const float*)d_in[0];
    const float* K = (const float*)d_in[1];
    const float* V = (const float*)d_in[2];
    const unsigned char* mask = (const unsigned char*)d_in[3];

    float* ctx  = (float*)d_out;                                 // [B,H,S,D] first
    float* attn = (float*)d_out + (size_t)BB * HH * SS * DD;     // then [B,H,S,S]

    dim3 grid(SS / QTILE, HH, BB);   // (128, 16, 2)
    dim3 block(NWAVES * 32);         // 4 wave32s
    size_t lds_bytes = (size_t)QTILE * FSTRIDE * sizeof(float)
                     + (size_t)QTILE * HSTRIDE * sizeof(_Float16);  // ~192 KB < 320 KB
    sdpa_fused_kernel<<<grid, block, lds_bytes, stream>>>(Q, K, V, mask, ctx, attn);
}
